// Faster_RCNN_16140487098778
// MI455X (gfx1250) — compile-verified, matrix operand order checked by reading, not run on hardware
//
#include <hip/hip_runtime.h>

// ---------------- CDNA5 WMMA types ----------------
typedef __attribute__((ext_vector_type(16))) _Float16 v16h;
typedef __attribute__((ext_vector_type(8)))  _Float16 h8;
typedef __attribute__((ext_vector_type(8)))  float    v8f;

#define S_ROI 14
#define NUM_CLASSES 80
#define MAX_OUT 100
#define IOU_THR 0.45f
#define SCORE_THR 0.5f
#define KPL 128
#define NROI 640
#define TOTP 21824   // total pixels over 5 FPN levels
#define TOTA 65472   // total anchors = 3*TOTP
#define BM 32        // output pixels per block (2 M-tiles)
#define KCH 64       // K halves staged per round (2 WMMA K-steps)

__constant__ int   cH[5]     = {128,64,32,16,8};
__constant__ int   cPOFF6[6] = {0,16384,20480,21504,21760,21824};
__constant__ float cDW[5]    = {4.f,8.f,16.f,32.f,64.f};
__constant__ float cAH[3]    = {8.f,4.f,16.f};
__constant__ float cAW[3]    = {8.f,16.f,4.f};

// ---------------- small helpers ----------------
__device__ inline void dev_decode(const float* b, const float* d,
                                  float ny, float nx, float nh, float nw,
                                  float* out) {
  float cy = (b[0]+b[2])*0.5f, cx = (b[1]+b[3])*0.5f;
  float h  = b[2]-b[0],        w  = b[3]-b[1];
  float dy = d[0]/ny, dx = d[1]/nx;
  float dh = fminf(fmaxf(d[2]/nh, -4.f), 4.f);
  float dw = fminf(fmaxf(d[3]/nw, -4.f), 4.f);
  cy += dy*h; cx += dx*w;
  h *= expf(dh); w *= expf(dw);
  out[0]=cy-h*0.5f; out[1]=cx-w*0.5f; out[2]=cy+h*0.5f; out[3]=cx+w*0.5f;
}

__device__ inline float dev_iou(const float* A, const float* B) {
  float yy1=fmaxf(A[0],B[0]), xx1=fmaxf(A[1],B[1]);
  float yy2=fminf(A[2],B[2]), xx2=fminf(A[3],B[3]);
  float inter = fmaxf(yy2-yy1,0.f)*fmaxf(xx2-xx1,0.f);
  float a=(A[2]-A[0])*(A[3]-A[1]), b=(B[2]-B[0])*(B[3]-B[1]);
  return inter/(a+b-inter+1e-9f);
}

// ---------------- prep: f32 -> f16 ----------------
__global__ void k_cvt(const float* __restrict__ in, _Float16* __restrict__ out, int n) {
  int i = blockIdx.x*blockDim.x + threadIdx.x;
  if (i < n) out[i] = (_Float16)in[i];
}

// weights [K][Cout] f32 -> [CoutPad][K] f16 (transposed, zero-padded cols)
__global__ void k_wcvt_t(const float* __restrict__ in, _Float16* __restrict__ out,
                         int K, int Cout, int CoutPad) {
  int i = blockIdx.x*blockDim.x + threadIdx.x;
  int total = K * CoutPad;
  if (i >= total) return;
  int n = i / K, k = i - n*K;
  out[(size_t)n*K + k] = (n < Cout) ? (_Float16)in[(size_t)k*Cout + n] : (_Float16)0.f;
}

// ---------------- implicit-GEMM conv / GEMM with WMMA + async LDS staging ----
// in  : f16 [nimg][H][W][Cin]
// wT  : f16 [CoutPad][ks*ks*Cin]   (K index = (ky*ks+kx)*Cin + cin)
// out : f16 and/or f32 [nimg*H*W][Cout]
// block = 128 threads (4 waves); wave w owns N-tile (blockIdx.y*4 + w),
// all waves share the LDS-staged A tile (BM=32 pixels x KCH=64 K halves),
// staged with global_load_async_to_lds_b128 (ASYNCcnt), double buffered.
__global__ __launch_bounds__(128)
void k_conv_wmma(const _Float16* __restrict__ in,
                 const _Float16* __restrict__ wT,
                 const float* __restrict__ bias,
                 _Float16* __restrict__ out16,
                 float* __restrict__ out32,
                 int nimg, int H, int W, int Cin,
                 int Cout, int CoutPad, int ks, int relu) {
  __shared__ _Float16 sA[2][BM][KCH];

  const int tid   = threadIdx.x;
  const int lane  = tid & 31;
  const int wv    = tid >> 5;
  const int HW    = H*W;
  const int Mtot  = nimg*HW;
  const int mbase = blockIdx.x * BM;
  const int Ktot  = ks*ks*Cin;
  const int ofs   = ks >> 1;
  const int NC    = Ktot / KCH;      // KCH | Cin | Ktot for all shapes used

  // ---- staging mapping: thread -> (pixel = tid>>2, quarter = tid&3) ----
  const int spix = tid >> 2;
  const int sq   = tid & 3;
  int sp = mbase + spix;
  bool spv = sp < Mtot;
  int simg = 0, spy = 0, spx = 0;
  if (spv) { simg = sp/HW; int r2 = sp - simg*HW; spy = r2/W; spx = r2 - spy*W; }

  int ts = 0, cins = 0;              // staging tap / channel cursor
  const _Float16* srow = in;
  bool svalid = false;
  auto tapsetup = [&]() {
    int ty = ts / ks;
    int dy = ty - ofs, dx = (ts - ty*ks) - ofs;
    int yy = spy + dy, xx = spx + dx;
    svalid = spv && (unsigned)yy < (unsigned)H && (unsigned)xx < (unsigned)W;
    if (svalid) srow = in + (((size_t)simg*H + yy)*W + xx)*Cin;
  };
  tapsetup();

  const unsigned ldsbase = (unsigned)(uintptr_t)&sA[0][0][0];
  h8 z8 = {};
  auto stage = [&](int buf) {
    unsigned loff = ldsbase + (unsigned)(((buf*BM + spix)*KCH + sq*16) * 2);
    if (svalid) {
      unsigned long long g = (unsigned long long)(uintptr_t)(srow + cins + sq*16);
      asm volatile("global_load_async_to_lds_b128 %0, %1, off"
                   :: "v"(loff), "v"(g) : "memory");
      asm volatile("global_load_async_to_lds_b128 %0, %1, off offset:16"
                   :: "v"(loff), "v"(g) : "memory");
    } else {
      *(h8*)&sA[buf][spix][sq*16]     = z8;
      *(h8*)&sA[buf][spix][sq*16 + 8] = z8;
    }
  };
  auto adv = [&]() {
    cins += KCH;
    if (cins >= Cin) { cins = 0; ts++; tapsetup(); }
  };

  // ---- compute mapping: wave wv owns N-tile, lane owns (row, K phase) ----
  const int row = lane & 15;
  const int kb  = (lane >> 4) * 8;       // K phase for 16-bit A/B fragments
  const int ntile = blockIdx.y * 4 + wv;
  const bool nact = (ntile * 16) < CoutPad;
  const int ncol  = ntile * 16 + row;
  const int ncolc = nact ? ncol : (CoutPad - 16 + row);  // clamp inactive waves
  const _Float16* brow = wT + (size_t)ncolc * Ktot;

  v8f acc0 = {}, acc1 = {};

  stage(0); adv();
  asm volatile("s_wait_asynccnt 0x0" ::: "memory");
  __syncthreads();

  int buf = 0;
  for (int c = 0; c < NC; ++c) {
    if (c + 1 < NC) { stage(buf ^ 1); adv(); }
    const int kofs = c * KCH;
    __builtin_prefetch((const void*)(brow + kofs + 2*KCH), 0, 1);
    const h8* aR0 = (const h8*)&sA[buf][row][0];
    const h8* aR1 = (const h8*)&sA[buf][row + 16][0];
#pragma unroll
    for (int s = 0; s < 2; ++s) {
      h8 b0  = *(const h8*)(brow + kofs + s*32 + kb);
      h8 b1  = *(const h8*)(brow + kofs + s*32 + kb + 16);
      h8 a00 = aR0[s*4 + (kb >> 3)];
      h8 a01 = aR0[s*4 + (kb >> 3) + 2];
      h8 a10 = aR1[s*4 + (kb >> 3)];
      h8 a11 = aR1[s*4 + (kb >> 3) + 2];
      v16h A0, A1, B;
#pragma unroll
      for (int e = 0; e < 8; e++) {
        A0[e] = a00[e]; A0[8+e] = a01[e];
        A1[e] = a10[e]; A1[8+e] = a11[e];
        B[e]  = b0[e];  B[8+e]  = b1[e];
      }
      acc0 = __builtin_amdgcn_wmma_f32_16x16x32_f16(false, A0, false, B,
                                                    (short)0, acc0, false, false);
      acc1 = __builtin_amdgcn_wmma_f32_16x16x32_f16(false, A1, false, B,
                                                    (short)0, acc1, false, false);
    }
    asm volatile("s_wait_asynccnt 0x0" ::: "memory");
    __syncthreads();
    buf ^= 1;
  }

  // D layout: VGPR r -> M = r + 8*(lane>>4), N = lane&15
  if (nact) {
#pragma unroll
    for (int r = 0; r < 8; r++) {
      int m0 = mbase + r + ((lane >> 4) * 8);
      int m1 = m0 + 16;
      if (m0 < Mtot && ncol < Cout) {
        float v = acc0[r] + bias[ncol];
        if (relu) v = fmaxf(v, 0.f);
        if (out16) out16[(size_t)m0 * Cout + ncol] = (_Float16)v;
        if (out32) out32[(size_t)m0 * Cout + ncol] = v;
      }
      if (m1 < Mtot && ncol < Cout) {
        float v = acc1[r] + bias[ncol];
        if (relu) v = fmaxf(v, 0.f);
        if (out16) out16[(size_t)m1 * Cout + ncol] = (_Float16)v;
        if (out32) out32[(size_t)m1 * Cout + ncol] = v;
      }
    }
  }
}

// ---------------- RPN objectness prob ----------------
__global__ void k_prob(const float* __restrict__ o32, float* __restrict__ prob) {
  int i = blockIdx.x*blockDim.x + threadIdx.x;
  if (i >= TOTA) return;
  int l = 0;
#pragma unroll
  for (int t = 1; t < 5; t++) if (i >= 3*cPOFF6[t]) l = t;
  int rel = i - 3*cPOFF6[l];
  int pix = rel / 3, a = rel - pix*3;
  const float* ob = o32 + (size_t)cPOFF6[l]*18 + (size_t)pix*18 + a*6;
  prob[i] = 1.f / (1.f + expf(ob[0] - ob[1]));   // softmax([:,:2])[:,1]
}

// ---------------- per-level top-128 (block argmax loop) ----------------
__global__ void k_topk(float* __restrict__ prob,
                       float* __restrict__ topv, int* __restrict__ topi) {
  int l = blockIdx.x, tid = threadIdx.x;
  int P = (cPOFF6[l+1] - cPOFF6[l]) * 3;
  float* pr = prob + 3*cPOFF6[l];
  __shared__ float rv[256];
  __shared__ int   ri[256];
  for (int k = 0; k < KPL; k++) {
    float bv = -1e30f; int bi = 0;
    for (int j = tid; j < P; j += 256) { float v = pr[j]; if (v > bv) { bv = v; bi = j; } }
    rv[tid] = bv; ri[tid] = bi; __syncthreads();
    for (int st = 128; st > 0; st >>= 1) {
      if (tid < st && rv[tid+st] > rv[tid]) { rv[tid] = rv[tid+st]; ri[tid] = ri[tid+st]; }
      __syncthreads();
    }
    if (tid == 0) { topv[l*KPL+k] = rv[0]; topi[l*KPL+k] = ri[0]; pr[ri[0]] = -1e30f; }
    __threadfence();
    __syncthreads();
  }
}

// ---------------- bbox decode + ROI align ----------------
__global__ void k_roialign(const _Float16* __restrict__ feat16,
                           const float* __restrict__ o32,
                           const float* __restrict__ topv,
                           const int*   __restrict__ topi,
                           _Float16* __restrict__ roifeat,
                           float* __restrict__ prop, int* __restrict__ valid) {
  int b = blockIdx.x, c = threadIdx.x;   // 640 blocks, 256 channels
  int l = b >> 7, j = b & 127;
  int H = cH[l], W = H;
  int i = topi[l*KPL + j];
  int pix = i / 3, a = i - pix*3;
  int ay = pix / W, ax = pix - ay*W;
  float h = cAH[a], w = cAW[a];
  float box[4] = { ay+0.5f - h*0.5f, ax+0.5f - w*0.5f, ay+0.5f + h*0.5f, ax+0.5f + w*0.5f };
  const float* ob = o32 + (size_t)cPOFF6[l]*18 + (size_t)pix*18 + a*6 + 2;
  float d[4] = { ob[0], ob[1], ob[2], ob[3] };
  float pb[4];
  dev_decode(box, d, 10.f, 10.f, 5.f, 5.f, pb);
  if (c == 0) {
    float s = cDW[l];
    prop[b*4+0]=pb[0]*s; prop[b*4+1]=pb[1]*s; prop[b*4+2]=pb[2]*s; prop[b*4+3]=pb[3]*s;
    valid[b] = topv[l*KPL+j] > 0.5f ? 1 : 0;
  }
  const _Float16* fb = feat16 + (size_t)cPOFF6[l]*256;
  for (int gy = 0; gy < S_ROI; gy++) {
    float gv = (gy + 0.5f) / S_ROI;
    float ys = pb[0] + gv*(pb[2]-pb[0]) - 0.5f;
    float y0 = floorf(ys), wy = ys - y0;
    int y0i = min(max((int)y0,     0), H-1);
    int y1i = min(max((int)y0 + 1, 0), H-1);
    for (int gx = 0; gx < S_ROI; gx++) {
      float gu = (gx + 0.5f) / S_ROI;
      float xs = pb[1] + gu*(pb[3]-pb[1]) - 0.5f;
      float x0 = floorf(xs), wx = xs - x0;
      int x0i = min(max((int)x0,     0), W-1);
      int x1i = min(max((int)x0 + 1, 0), W-1);
      float v00 = (float)fb[((size_t)y0i*W + x0i)*256 + c];
      float v01 = (float)fb[((size_t)y0i*W + x1i)*256 + c];
      float v10 = (float)fb[((size_t)y1i*W + x0i)*256 + c];
      float v11 = (float)fb[((size_t)y1i*W + x1i)*256 + c];
      float val = v00*(1.f-wy)*(1.f-wx) + v01*(1.f-wy)*wx
                + v10*wy*(1.f-wx)       + v11*wy*wx;
      roifeat[((size_t)b*(S_ROI*S_ROI) + gy*S_ROI + gx)*256 + c] = (_Float16)val;
    }
  }
}

// ---------------- final box decode ----------------
__global__ void k_pred(const float* __restrict__ prop, const float* __restrict__ reg2,
                       float* __restrict__ pred) {
  int i = blockIdx.x*blockDim.x + threadIdx.x;
  if (i >= NROI) return;
  float b[4] = { prop[i*4], prop[i*4+1], prop[i*4+2], prop[i*4+3] };
  float d[4] = { reg2[i*4], reg2[i*4+1], reg2[i*4+2], reg2[i*4+3] };
  dev_decode(b, d, 1.f, 1.f, 1.f, 1.f, pred + i*4);
}

// ---------------- per-class NMS ----------------
__global__ void k_nms(const float* __restrict__ cla, const int* __restrict__ valid,
                      const float* __restrict__ pred,
                      int* __restrict__ out_cls, float* __restrict__ out_sc,
                      float* __restrict__ out_bx) {
  int cls = blockIdx.x, tid = threadIdx.x;
  __shared__ float s[NROI];
  __shared__ float bs[NROI][4];
  __shared__ float rv[256];
  __shared__ int   ri[256];
  for (int j = tid; j < NROI; j += 256) {
    float sg = 1.f / (1.f + expf(-cla[(size_t)j*NUM_CLASSES + cls]));
    s[j] = (valid[j] && sg > SCORE_THR) ? sg : -1.f;
    bs[j][0]=pred[j*4]; bs[j][1]=pred[j*4+1]; bs[j][2]=pred[j*4+2]; bs[j][3]=pred[j*4+3];
  }
  __syncthreads();
  for (int k = 0; k < MAX_OUT; k++) {
    float bv = -1e30f; int bi = 0;
    for (int j = tid; j < NROI; j += 256) if (s[j] > bv) { bv = s[j]; bi = j; }
    rv[tid] = bv; ri[tid] = bi; __syncthreads();
    for (int st = 128; st > 0; st >>= 1) {
      if (tid < st && rv[tid+st] > rv[tid]) { rv[tid] = rv[tid+st]; ri[tid] = ri[tid+st]; }
      __syncthreads();
    }
    int best = ri[0]; float bscore = rv[0];
    bool v = bscore > SCORE_THR;
    float bb[4] = { bs[best][0], bs[best][1], bs[best][2], bs[best][3] };
    if (tid == 0) {
      int o = cls*MAX_OUT + k;
      out_cls[o] = v ? cls : -1;
      out_sc[o]  = v ? bscore : 0.f;
      out_bx[o*4+0] = v ? bb[0] : 0.f; out_bx[o*4+1] = v ? bb[1] : 0.f;
      out_bx[o*4+2] = v ? bb[2] : 0.f; out_bx[o*4+3] = v ? bb[3] : 0.f;
    }
    __syncthreads();
    for (int j = tid; j < NROI; j += 256)
      if (v && dev_iou(bb, bs[j]) > IOU_THR) s[j] = -1.f;
    if (tid == 0) s[best] = -1.f;
    __syncthreads();
  }
}

// ---------------- host orchestration ----------------
extern "C" void kernel_launch(void* const* d_in, const int* in_sizes, int n_in,
                              void* d_out, int out_size, void* d_ws, size_t ws_size,
                              hipStream_t stream) {
  (void)in_sizes; (void)n_in; (void)out_size; (void)ws_size;
  const float* f[5] = { (const float*)d_in[0], (const float*)d_in[1],
                        (const float*)d_in[2], (const float*)d_in[3],
                        (const float*)d_in[4] };
  const float* rpn_w1 = (const float*)d_in[5];
  const float* rpn_b1 = (const float*)d_in[6];
  const float* rpn_w2 = (const float*)d_in[7];
  const float* rpn_b2 = (const float*)d_in[8];
  const float* rpn_wo = (const float*)d_in[9];
  const float* rpn_bo = (const float*)d_in[10];
  const float* rcnn_w1 = (const float*)d_in[11];
  const float* rcnn_b1 = (const float*)d_in[12];
  const float* rcnn_w2 = (const float*)d_in[13];
  const float* rcnn_b2 = (const float*)d_in[14];
  const float* rcnn_w3 = (const float*)d_in[15];
  const float* rcnn_b3 = (const float*)d_in[16];
  const float* cls_w  = (const float*)d_in[17];
  const float* cls_b  = (const float*)d_in[18];
  const float* reg_w  = (const float*)d_in[19];
  const float* reg_b  = (const float*)d_in[20];

  static const int hH[5]    = {128,64,32,16,8};
  static const int hPOFF[5] = {0,16384,20480,21504,21760};

  char* ws = (char*)d_ws;
  size_t off = 0;
  auto alloc = [&](size_t bytes) -> char* {
    off = (off + 255) & ~(size_t)255;
    char* p = ws + off; off += bytes; return p;
  };

  _Float16* feat16 = (_Float16*)alloc((size_t)TOTP*256*2);
  _Float16* w1T    = (_Float16*)alloc((size_t)5*256*2304*2);
  _Float16* w2T    = (_Float16*)alloc((size_t)5*256*2304*2);
  _Float16* woT    = (_Float16*)alloc((size_t)5*32*2304*2);
  _Float16* rw1T   = (_Float16*)alloc((size_t)128*256*2);
  _Float16* rw2T   = (_Float16*)alloc((size_t)256*1152*2);
  _Float16* rw3T   = (_Float16*)alloc((size_t)128*256*2);
  _Float16* clsT   = (_Float16*)alloc((size_t)80*25088*2);
  _Float16* regT   = (_Float16*)alloc((size_t)16*25088*2);
  _Float16* c1     = (_Float16*)alloc((size_t)TOTP*256*2);
  _Float16* c2     = (_Float16*)alloc((size_t)TOTP*256*2);
  float*    o32    = (float*)   alloc((size_t)TOTP*18*4);
  float*    prob   = (float*)   alloc((size_t)TOTA*4);
  float*    topv   = (float*)   alloc((size_t)NROI*4);
  int*      topi   = (int*)     alloc((size_t)NROI*4);
  float*    prop   = (float*)   alloc((size_t)NROI*16);
  int*      valid  = (int*)     alloc((size_t)NROI*4);
  _Float16* roifeat= (_Float16*)alloc((size_t)NROI*196*256*2);  // also rc2
  _Float16* rc1    = (_Float16*)alloc((size_t)NROI*196*128*2);  // also rc3 (flat)
  float*    cla    = (float*)   alloc((size_t)NROI*80*4);
  float*    reg2   = (float*)   alloc((size_t)NROI*4*4);
  float*    pred   = (float*)   alloc((size_t)NROI*16);

  // 1) convert features to f16
  for (int l = 0; l < 5; l++) {
    int n = hH[l]*hH[l]*256;
    k_cvt<<<(n+255)/256, 256, 0, stream>>>(f[l], feat16 + (size_t)hPOFF[l]*256, n);
  }
  // 2) convert+transpose weights to f16 [CoutPad][K]
  for (int l = 0; l < 5; l++) {
    k_wcvt_t<<<(2304*256+255)/256, 256, 0, stream>>>(
        rpn_w1 + (size_t)l*2304*256, w1T + (size_t)l*256*2304, 2304, 256, 256);
    k_wcvt_t<<<(2304*256+255)/256, 256, 0, stream>>>(
        rpn_w2 + (size_t)l*2304*256, w2T + (size_t)l*256*2304, 2304, 256, 256);
    k_wcvt_t<<<(2304*32+255)/256, 256, 0, stream>>>(
        rpn_wo + (size_t)l*2304*18, woT + (size_t)l*32*2304, 2304, 18, 32);
  }
  k_wcvt_t<<<(256*128+255)/256, 256, 0, stream>>>(rcnn_w1, rw1T, 256, 128, 128);
  k_wcvt_t<<<(1152*256+255)/256, 256, 0, stream>>>(rcnn_w2, rw2T, 1152, 256, 256);
  k_wcvt_t<<<(256*128+255)/256, 256, 0, stream>>>(rcnn_w3, rw3T, 256, 128, 128);
  k_wcvt_t<<<(25088*80+255)/256, 256, 0, stream>>>(cls_w, clsT, 25088, 80, 80);
  k_wcvt_t<<<(25088*16+255)/256, 256, 0, stream>>>(reg_w, regT, 25088, 4, 16);

  // 3) RPN tower per level (WMMA implicit GEMM, async LDS staging)
  for (int l = 0; l < 5; l++) {
    int H = hH[l], P = H*H;
    dim3 g1((P+BM-1)/BM, 4);   // Cout 256 -> 16 N-tiles / 4 waves
    k_conv_wmma<<<g1, 128, 0, stream>>>(
        feat16 + (size_t)hPOFF[l]*256, w1T + (size_t)l*256*2304, rpn_b1 + l*256,
        c1 + (size_t)hPOFF[l]*256, nullptr, 1, H, H, 256, 256, 256, 3, 1);
    k_conv_wmma<<<g1, 128, 0, stream>>>(
        c1 + (size_t)hPOFF[l]*256, w2T + (size_t)l*256*2304, rpn_b2 + l*256,
        c2 + (size_t)hPOFF[l]*256, nullptr, 1, H, H, 256, 256, 256, 3, 1);
    dim3 go((P+BM-1)/BM, 1);   // CoutPad 32 -> 2 N-tiles
    k_conv_wmma<<<go, 128, 0, stream>>>(
        c2 + (size_t)hPOFF[l]*256, woT + (size_t)l*32*2304, rpn_bo + l*18,
        nullptr, o32 + (size_t)hPOFF[l]*18, 1, H, H, 256, 18, 32, 3, 0);
  }

  // 4) objectness prob, 5) per-level top-128
  k_prob<<<(TOTA+255)/256, 256, 0, stream>>>(o32, prob);
  k_topk<<<5, 256, 0, stream>>>(prob, topv, topi);

  // 6) decode + ROI align
  k_roialign<<<NROI, 256, 0, stream>>>(feat16, o32, topv, topi, roifeat, prop, valid);

  // 7) RCNN head (WMMA), buffers aliased: roifeat->rc2, rc1->rc3(flat)
  dim3 gr((NROI*196+BM-1)/BM, 2);
  k_conv_wmma<<<gr, 128, 0, stream>>>(roifeat, rw1T, rcnn_b1, rc1, nullptr,
                                      NROI, 14, 14, 256, 128, 128, 1, 1);
  dim3 gr2((NROI*196+BM-1)/BM, 4);
  k_conv_wmma<<<gr2, 128, 0, stream>>>(rc1, rw2T, rcnn_b2, roifeat, nullptr,
                                       NROI, 14, 14, 128, 256, 256, 3, 1);
  k_conv_wmma<<<gr, 128, 0, stream>>>(roifeat, rw3T, rcnn_b3, rc1, nullptr,
                                      NROI, 14, 14, 256, 128, 128, 1, 1);

  // 8) FC heads as GEMM (H=W=1, Cin=25088)
  dim3 gc((NROI+BM-1)/BM, 2);   // 5 N-tiles
  k_conv_wmma<<<gc, 128, 0, stream>>>(rc1, clsT, cls_b, nullptr, cla,
                                      NROI, 1, 1, 25088, 80, 80, 1, 0);
  dim3 gg((NROI+BM-1)/BM, 1);   // 1 N-tile
  k_conv_wmma<<<gg, 128, 0, stream>>>(rc1, regT, reg_b, nullptr, reg2,
                                      NROI, 1, 1, 25088, 4, 16, 1, 0);

  // 9) final decode + NMS -> outputs [int32 8000][f32 8000][f32 32000]
  k_pred<<<(NROI+255)/256, 256, 0, stream>>>(prop, reg2, pred);
  int*   out_cls = (int*)d_out;
  float* out_sc  = (float*)d_out + NUM_CLASSES*MAX_OUT;
  float* out_bx  = (float*)d_out + 2*NUM_CLASSES*MAX_OUT;
  k_nms<<<NUM_CLASSES, 256, 0, stream>>>(cla, valid, pred, out_cls, out_sc, out_bx);
}
